// CoarseQuantizer_45157286150849
// MI455X (gfx1250) — matrix operand order repaired and checked by reading, not compiled, not run on hardware
//
#include <hip/hip_runtime.h>
#include <hip/hip_bf16.h>

typedef __attribute__((ext_vector_type(16))) _Float16 v16h;
typedef __attribute__((ext_vector_type(8)))  _Float16 v8h;
typedef __attribute__((ext_vector_type(8)))  float    v8f;

#define NUM_CODE   512
#define CODE_DIM   64
#define WAVES      4
#define ROWS_PW    64            // tokens per wave (4 M-tiles of 16)
#define TOK_PB     (WAVES * ROWS_PW)   // 256 tokens per block
#define XS_STRIDE  72            // halves per row in LDS (144B pitch, 16B aligned)

// ---------------------------------------------------------------------------
// Kernel 1: codebook prep. 512 blocks x 64 threads (block = one code row).
//   - inverse L2 norm per code -> rnorm[]
//   - f16 normalized codebook  -> cbh[]   (WMMA B operand source, L2-hot)
//   - raw codebook passthrough -> out_cb
//   - zero counts / loss accumulator (deterministic each launch)
// ---------------------------------------------------------------------------
__global__ __launch_bounds__(64) void vq_prep(const float* __restrict__ cb,
                                              float* __restrict__ out_cb,
                                              _Float16* __restrict__ cbh,
                                              float* __restrict__ rnorm,
                                              unsigned* __restrict__ counts,
                                              float* __restrict__ loss_acc) {
    __shared__ float s[64];
    __shared__ float srn;
    const int n = blockIdx.x;
    const int d = threadIdx.x;
    const float v = cb[n * CODE_DIM + d];
    out_cb[n * CODE_DIM + d] = v;
    s[d] = v * v;
    __syncthreads();
    #pragma unroll
    for (int off = 32; off > 0; off >>= 1) {
        if (d < off) s[d] += s[d + off];
        __syncthreads();
    }
    if (d == 0) {
        const float rn = 1.0f / fmaxf(sqrtf(s[0]), 1e-12f);
        srn = rn;
        rnorm[n] = rn;
        counts[n] = 0u;
        if (n == 0) *loss_acc = 0.0f;
    }
    __syncthreads();
    cbh[n * CODE_DIM + d] = (_Float16)(v * srn);
}

// ---------------------------------------------------------------------------
// Kernel 2: fused normalize + WMMA similarity + argmax + gather + stats.
// 128 threads = 4 waves; each wave handles 64 tokens (4 x 16-row M-tiles).
// B fragments are software-pipelined: the loads for N-tile nt+1 are issued
// before the 8 WMMAs of N-tile nt so L2 latency overlaps XDL compute.
// ---------------------------------------------------------------------------
__global__ __launch_bounds__(128) void vq_main(const float* __restrict__ x,
                                               const float* __restrict__ cb,
                                               const _Float16* __restrict__ cbh,
                                               const float* __restrict__ rnorm,
                                               unsigned* __restrict__ counts,
                                               float* __restrict__ loss_acc,
                                               float* __restrict__ out_q,
                                               float* __restrict__ out_idx) {
    __shared__ _Float16 xs[WAVES * ROWS_PW * XS_STRIDE];   // 36 KB
    __shared__ int      idxb[WAVES * ROWS_PW];             // 1 KB

    const int tid  = threadIdx.x;
    const int wave = tid >> 5;
    const int lane = tid & 31;
    const long tokBase = (long)blockIdx.x * TOK_PB + (long)wave * ROWS_PW;
    _Float16* xw = xs + wave * ROWS_PW * XS_STRIDE;

    // ---- stage: normalize 2 token rows per lane, write f16 to LDS ----
    #pragma unroll
    for (int rr = 0; rr < 2; ++rr) {
        const int row = lane * 2 + rr;
        const float4* src = (const float4*)(x + (tokBase + row) * CODE_DIM);
        float4 v[16];
        float ss = 0.0f;
        #pragma unroll
        for (int i = 0; i < 16; ++i) {
            v[i] = src[i];
            ss += v[i].x * v[i].x + v[i].y * v[i].y + v[i].z * v[i].z + v[i].w * v[i].w;
        }
        const float rn = 1.0f / fmaxf(sqrtf(ss), 1e-12f);
        _Float16* dst = xw + row * XS_STRIDE;
        #pragma unroll
        for (int i = 0; i < 16; ++i) {
            dst[i * 4 + 0] = (_Float16)(v[i].x * rn);
            dst[i * 4 + 1] = (_Float16)(v[i].y * rn);
            dst[i * 4 + 2] = (_Float16)(v[i].z * rn);
            dst[i * 4 + 3] = (_Float16)(v[i].w * rn);
        }
    }
    __syncthreads();

    // ---- A fragments: 16x32 A, contiguous k order (k = 16h + s) ----
    // lane<16 holds K {0..15}, lane>=16 holds K {16..31} (+32 for f=1)
    const int kgrp = (lane >> 4) * 16;
    v16h A[4][2];
    #pragma unroll
    for (int mt = 0; mt < 4; ++mt) {
        const _Float16* rp = xw + (mt * 16 + (lane & 15)) * XS_STRIDE;
        #pragma unroll
        for (int f = 0; f < 2; ++f) {
            v8h lo = *(const v8h*)(rp + f * 32 + kgrp);
            v8h hi = *(const v8h*)(rp + f * 32 + kgrp + 8);
            A[mt][f] = __builtin_shufflevector(lo, hi, 0, 1, 2, 3, 4, 5, 6, 7,
                                               8, 9, 10, 11, 12, 13, 14, 15);
        }
    }

    // ---- WMMA similarity loop over 32 N-tiles with fused argmax ----
    float best[4][8];
    int   bn[4][8];
    #pragma unroll
    for (int mt = 0; mt < 4; ++mt)
        #pragma unroll
        for (int r = 0; r < 8; ++r) { best[mt][r] = -3.0e38f; bn[mt][r] = 0; }

    const int nlane = lane & 15;
    const int bsel  = (lane >> 4) * 16;   // B 32x16 layout: lane half selects K group
    const _Float16* bbase = cbh + nlane * CODE_DIM + bsel;

    // prologue: B fragments for nt = 0
    v16h b0 = *(const v16h*)(bbase);
    v16h b1 = *(const v16h*)(bbase + 32);

    for (int nt = 0; nt < 32; ++nt) {
        // issue next tile's B loads first (last iteration reloads tile 31)
        const int nnext = (nt < 31) ? (nt + 1) : 31;
        const _Float16* np = bbase + nnext * (16 * CODE_DIM);
        v16h nb0 = *(const v16h*)np;
        v16h nb1 = *(const v16h*)(np + 32);

        const int n = nt * 16 + nlane;
        #pragma unroll
        for (int mt = 0; mt < 4; ++mt) {
            v8f c = {};
            c = __builtin_amdgcn_wmma_f32_16x16x32_f16(false, A[mt][0], false, b0,
                                                       (short)0, c, false, false);
            c = __builtin_amdgcn_wmma_f32_16x16x32_f16(false, A[mt][1], false, b1,
                                                       (short)0, c, false, false);
            #pragma unroll
            for (int r = 0; r < 8; ++r) {
                const float cv = c[r];
                if (cv > best[mt][r]) { best[mt][r] = cv; bn[mt][r] = n; }
            }
        }
        b0 = nb0;
        b1 = nb1;
    }

    // ---- per-row argmax across the 16 lanes of each half (C layout) ----
    #pragma unroll
    for (int mt = 0; mt < 4; ++mt) {
        #pragma unroll
        for (int r = 0; r < 8; ++r) {
            float v = best[mt][r];
            int   n = bn[mt][r];
            #pragma unroll
            for (int m = 1; m < 16; m <<= 1) {
                const float ov = __shfl_xor(v, m, 32);
                const int   on = __shfl_xor(n, m, 32);
                if (ov > v || (ov == v && on < n)) { v = ov; n = on; }
            }
            if ((lane & 15) == 0) {
                const int row = mt * 16 + r + (lane >> 4) * 8;
                idxb[wave * ROWS_PW + row] = n;
                out_idx[tokBase + row] = (float)n;
                atomicAdd(&counts[n], 1u);
            }
        }
    }
    __syncthreads();

    // ---- gather normalized code, write quantized, accumulate loss ----
    float lsum = 0.0f;
    #pragma unroll
    for (int mt = 0; mt < 4; ++mt) {
        const int row = mt * 16 + (lane >> 1);
        const int idx = idxb[wave * ROWS_PW + row];
        const float rn = rnorm[idx];
        const int colb = (lane & 1) * 32;
        const float*    crow = cb + idx * CODE_DIM + colb;
        const _Float16* xrow = xw + row * XS_STRIDE + colb;
        float*          qrow = out_q + (tokBase + row) * CODE_DIM + colb;
        #pragma unroll
        for (int cix = 0; cix < 32; ++cix) {
            const float q  = crow[cix] * rn;
            const float xv = (float)xrow[cix];
            qrow[cix] = q;
            const float d = q - xv;
            lsum += d * d;
        }
    }
    #pragma unroll
    for (int m = 1; m < 32; m <<= 1) lsum += __shfl_xor(lsum, m, 32);
    if (lane == 0) atomicAdd(loss_acc, lsum);
}

// ---------------------------------------------------------------------------
// Kernel 3: perplexity + loss scalars. 1 block x 512 threads.
// ---------------------------------------------------------------------------
__global__ __launch_bounds__(512) void vq_final(const unsigned* __restrict__ counts,
                                                const float* __restrict__ loss_acc,
                                                float* __restrict__ out,
                                                int nTok, long perpOff) {
    __shared__ float s[NUM_CODE];
    const int t = threadIdx.x;
    const float p = (float)counts[t] / (float)nTok;
    s[t] = p * logf(p + 1e-10f);
    __syncthreads();
    #pragma unroll
    for (int off = 256; off > 0; off >>= 1) {
        if (t < off) s[t] += s[t + off];
        __syncthreads();
    }
    if (t == 0) {
        out[perpOff] = expf(-s[0]);
        out[0] = 1.25f * (*loss_acc) / ((float)nTok * (float)CODE_DIM);
    }
}

// ---------------------------------------------------------------------------
extern "C" void kernel_launch(void* const* d_in, const int* in_sizes, int n_in,
                              void* d_out, int out_size, void* d_ws, size_t ws_size,
                              hipStream_t stream) {
    const float* x  = (const float*)d_in[0];   // [B,C,T,D] f32
    const float* cb = (const float*)d_in[1];   // [512,64] f32
    const int xN  = in_sizes[0];               // 8388608
    const int cbN = in_sizes[1];               // 32768
    const int M   = xN / CODE_DIM;             // 131072 tokens

    float* out = (float*)d_out;
    const long O_Q    = 1;
    const long O_PERP = 1 + (long)xN;
    const long O_CB   = O_PERP + 1;
    const long O_IDX  = O_CB + (long)cbN;

    // workspace layout
    unsigned* counts   = (unsigned*)d_ws;                          // 512 * u32
    float*    loss_acc = (float*)((char*)d_ws + 2048);             // 1 float
    float*    rnorm    = (float*)((char*)d_ws + 2560);             // 512 floats
    _Float16* cbh      = (_Float16*)((char*)d_ws + 4608);          // 512*64 f16 (32B aligned)

    vq_prep<<<NUM_CODE, CODE_DIM, 0, stream>>>(cb, out + O_CB, cbh, rnorm,
                                               counts, loss_acc);
    vq_main<<<M / TOK_PB, WAVES * 32, 0, stream>>>(x, cb, cbh, rnorm, counts,
                                                   loss_acc, out + O_Q, out + O_IDX);
    vq_final<<<1, NUM_CODE, 0, stream>>>(counts, loss_acc, out, M, O_PERP);
}
